// baselineGRU_15985868276503
// MI455X (gfx1250) — matrix operand order repaired and checked by reading, not compiled, not run on hardware
//
#include <hip/hip_runtime.h>

// ---------------------------------------------------------------------------
// 2-layer GRU (H=32, in=4), B=4096, S=512, fused per-timestep, + FC head.
//
// 256 workgroups x 64 threads (2 waves). Workgroup owns 16 batch rows; each
// wave owns 16 of the 32 hidden columns (3 of 6 gate N-tiles: r/z/n slices).
// Per step, per wave (C accumulators start at inline 0; biases added in the
// elementwise phase):
//   GhA = hA  @ W_hh0^T    -> 3x v_wmma_f32_16x16x32_f16
//   GxA = x_t @ W_ih0^T    -> 3x v_wmma_f32_16x16x4_f32 (computed in the
//                             previous step's barrier-overlap slot)
//   elementwise -> hA_new; LDS exchange -> A-format f16 operand
//   GxB = hA_new @ W_ih1^T -> 3x f16 WMMA
//   GhB = hB     @ W_hh1^T -> 3x f16 WMMA (overlapped with barrier A)
//   elementwise -> hB_new; LDS exchange
// Split s_barrier_signal/s_barrier_wait hide barrier+LDS latency behind
// independent WMMAs and the software-pipelined x load.
// ---------------------------------------------------------------------------

typedef __attribute__((ext_vector_type(16))) _Float16 v16h;
typedef __attribute__((ext_vector_type(8)))  float    v8f;
typedef __attribute__((ext_vector_type(2)))  float    v2f;

#define HID   32
#define SEQ   512
#define BATCH 4096
#define LDS_STRIDE 40   // halves per row: 32 data + 8 pad (80B, conflict-free b128)

__device__ __forceinline__ float fast_sig(float x) {
    float e = __builtin_amdgcn_exp2f(x * -1.4426950408889634f);
    return __builtin_amdgcn_rcpf(1.0f + e);
}
__device__ __forceinline__ float fast_tanh(float x) {
    float e = __builtin_amdgcn_exp2f(x * -2.8853900817779268f);
    return 2.0f * __builtin_amdgcn_rcpf(1.0f + e) - 1.0f;
}
__device__ __forceinline__ v8f wmma_h0(v16h a, v16h b) {
    v8f z = {};
    return __builtin_amdgcn_wmma_f32_16x16x32_f16(false, a, false, b,
                                                  (short)0, z, false, false);
}
__device__ __forceinline__ v8f wmma_s0(v2f a, v2f b) {
    v8f z = {};
    return __builtin_amdgcn_wmma_f32_16x16x4_f32(false, a, false, b,
                                                 (short)0, z, false, false);
}

// Split workgroup barrier: stores drained, signal arrival; wait separately so
// independent WMMAs can execute in the signal->wait window.
#define BAR_SIGNAL() asm volatile("s_wait_dscnt 0x0\n\ts_barrier_signal -1" ::: "memory")
#define BAR_WAIT()   asm volatile("s_barrier_wait -1" ::: "memory")

union AOp { uint4 q[2]; v16h a; };

__global__ __launch_bounds__(64) void gru_seq_kernel(
    const float* __restrict__ x,
    const float* __restrict__ Wih0, const float* __restrict__ Whh0,
    const float* __restrict__ bih0, const float* __restrict__ bhh0,
    const float* __restrict__ Wih1, const float* __restrict__ Whh1,
    const float* __restrict__ bih1, const float* __restrict__ bhh1,
    float* __restrict__ h_last)
{
    __shared__ __align__(16) _Float16 ldsA[16 * LDS_STRIDE];
    __shared__ __align__(16) _Float16 ldsB[16 * LDS_STRIDE];

    const int lane = threadIdx.x & 31;
    const int w    = threadIdx.x >> 5;   // wave id: column half 0/1
    const int ln   = lane & 15;          // N within tile (and M for A loads)
    const int hi   = lane >> 4;          // lane half
    const int b0   = blockIdx.x * 16;

    // ---- Load weights into resident B-format register tiles ---------------
    // f16 B (32x16): lane holds N=ln; VGPR v holds K = (2v+16*hi, 2v+1+16*hi)
    // fp32 B (4x16): lane holds N=ln; VGPR v holds K = v + 2*hi
    v16h BhhA[3], BihB[3], BhhB[3];
    v2f  BihA[3];
#pragma unroll
    for (int gi = 0; gi < 3; ++gi) {
        const int t = gi * 2 + w;        // gate N-tile index (0..5)
        const int g = t * 16 + ln;       // gate row 0..95
#pragma unroll
        for (int v = 0; v < 8; ++v) {
            const int k = 2 * v + 16 * hi;
            float2 w0 = *(const float2*)(Whh0 + g * HID + k);
            float2 w1 = *(const float2*)(Wih1 + g * HID + k);
            float2 w2 = *(const float2*)(Whh1 + g * HID + k);
            BhhA[gi][2*v] = (_Float16)w0.x;  BhhA[gi][2*v+1] = (_Float16)w0.y;
            BihB[gi][2*v] = (_Float16)w1.x;  BihB[gi][2*v+1] = (_Float16)w1.y;
            BhhB[gi][2*v] = (_Float16)w2.x;  BhhB[gi][2*v+1] = (_Float16)w2.y;
        }
        BihA[gi][0] = Wih0[g * 4 + 2 * hi + 0];
        BihA[gi][1] = Wih0[g * 4 + 2 * hi + 1];
    }
    // Per-lane bias scalars (gate rows for this wave's r/z/n slices).
    const int g_r = 16 * w + ln, g_z = 32 + g_r, g_n = 64 + g_r;
    const float bsrA = bih0[g_r] + bhh0[g_r];
    const float bszA = bih0[g_z] + bhh0[g_z];
    const float bxnA = bih0[g_n], bhnA = bhh0[g_n];
    const float bsrB = bih1[g_r] + bhh1[g_r];
    const float bszB = bih1[g_z] + bhh1[g_z];
    const float bxnB = bih1[g_n], bhnB = bhh1[g_n];

    // ---- State ------------------------------------------------------------
    v16h ahA, ahB;                       // A-format f16 hidden states
#pragma unroll
    for (int i = 0; i < 16; ++i) { ahA[i] = (_Float16)0.f; ahB[i] = (_Float16)0.f; }
    float hAd[8], hBd[8];                // D-layout f32 copies (this wave's cols)
#pragma unroll
    for (int k = 0; k < 8; ++k) { hAd[k] = 0.f; hBd[k] = 0.f; }

    const float* xrow = x + (size_t)(b0 + ln) * (SEQ * 4) + 2 * hi;
    const int st_off = 8 * hi * LDS_STRIDE + 16 * w + ln;  // LDS store base
    const int ld_off = ln * LDS_STRIDE + hi * 16;          // A-format load base

    // Software pipeline: GxA for t=0 computed before the loop.
    float2 xv = *(const float2*)(xrow);
    v8f GxA[3];
    {
        v2f ax; ax[0] = xv.x; ax[1] = xv.y;
#pragma unroll
        for (int gi = 0; gi < 3; ++gi) GxA[gi] = wmma_s0(ax, BihA[gi]);
    }

    for (int t = 0; t < SEQ; ++t) {
        // ---- Layer 0: recurrent matmul + gates ----
        v8f GhA[3];
#pragma unroll
        for (int gi = 0; gi < 3; ++gi) GhA[gi] = wmma_h0(ahA, BhhA[gi]);
#pragma unroll
        for (int k = 0; k < 8; ++k) {
            float r = fast_sig(GxA[0][k] + GhA[0][k] + bsrA);
            float z = fast_sig(GxA[1][k] + GhA[1][k] + bszA);
            float n = fast_tanh(GxA[2][k] + bxnA + r * (GhA[2][k] + bhnA));
            float hn = n + z * (hAd[k] - n);
            hAd[k] = hn;
            ldsA[k * LDS_STRIDE + st_off] = (_Float16)hn;
        }
        BAR_SIGNAL();
        // Overlap slot A: GhB depends only on previous-step ahB; also issue
        // next-step x load + long-range prefetch.
        v8f GhB[3];
#pragma unroll
        for (int gi = 0; gi < 3; ++gi) GhB[gi] = wmma_h0(ahB, BhhB[gi]);
        const int tn = (t + 1 < SEQ) ? t + 1 : SEQ - 1;
        float2 xv2 = *(const float2*)(xrow + tn * 4);
        __builtin_prefetch(xrow + (t + 16) * 4, 0, 0);
        BAR_WAIT();
        AOp a1;
        a1.q[0] = *(const uint4*)(ldsA + ld_off);
        a1.q[1] = *(const uint4*)(ldsA + ld_off + 8);
        ahA = a1.a;

        // ---- Layer 1 ----
        v8f GxB[3];
#pragma unroll
        for (int gi = 0; gi < 3; ++gi) GxB[gi] = wmma_h0(ahA, BihB[gi]);
#pragma unroll
        for (int k = 0; k < 8; ++k) {
            float r = fast_sig(GxB[0][k] + GhB[0][k] + bsrB);
            float z = fast_sig(GxB[1][k] + GhB[1][k] + bszB);
            float n = fast_tanh(GxB[2][k] + bxnB + r * (GhB[2][k] + bhnB));
            float hn = n + z * (hBd[k] - n);
            hBd[k] = hn;
            ldsB[k * LDS_STRIDE + st_off] = (_Float16)hn;
        }
        BAR_SIGNAL();
        // Overlap slot B: next-step GxA from the pipelined x value.
        {
            v2f ax; ax[0] = xv2.x; ax[1] = xv2.y;
#pragma unroll
            for (int gi = 0; gi < 3; ++gi) GxA[gi] = wmma_s0(ax, BihA[gi]);
        }
        BAR_WAIT();
        AOp a2;
        a2.q[0] = *(const uint4*)(ldsB + ld_off);
        a2.q[1] = *(const uint4*)(ldsB + ld_off + 8);
        ahB = a2.a;
        xv = xv2;
    }

    // ---- Write final layer-1 hidden state (fp32, D-layout regs) -----------
#pragma unroll
    for (int k = 0; k < 8; ++k) {
        h_last[(size_t)(b0 + k + 8 * hi) * HID + 16 * w + ln] = hBd[k];
    }
}

// ---------------------------------------------------------------------------
// FC head: out = (h_last @ Wfc2^T + b_fc2) @ Wfc^T + b_fc   (4096 x 1)
// ---------------------------------------------------------------------------
__global__ __launch_bounds__(256) void fc_head_kernel(
    const float* __restrict__ hl,
    const float* __restrict__ Wfc2, const float* __restrict__ bfc2,
    const float* __restrict__ Wfc,  const float* __restrict__ bfc,
    float* __restrict__ out)
{
    __shared__ float sW[HID * HID];
    __shared__ float sb2[HID];
    __shared__ float sw1[HID];
    const int tid = threadIdx.x;
    for (int i = tid; i < HID * HID; i += 256) sW[i] = Wfc2[i];
    if (tid < HID) { sb2[tid] = bfc2[tid]; sw1[tid] = Wfc[tid]; }
    __syncthreads();

    const int b = blockIdx.x * 256 + tid;
    float h[HID];
#pragma unroll
    for (int c = 0; c < HID; ++c) h[c] = hl[(size_t)b * HID + c];
    float acc = bfc[0];
#pragma unroll 4
    for (int g = 0; g < HID; ++g) {
        float o = sb2[g];
#pragma unroll
        for (int c = 0; c < HID; ++c) o += sW[g * HID + c] * h[c];
        acc += o * sw1[g];
    }
    out[b] = acc;
}

extern "C" void kernel_launch(void* const* d_in, const int* in_sizes, int n_in,
                              void* d_out, int out_size, void* d_ws, size_t ws_size,
                              hipStream_t stream) {
    const float* x    = (const float*)d_in[0];
    const float* Wih0 = (const float*)d_in[1];
    const float* Whh0 = (const float*)d_in[2];
    const float* bih0 = (const float*)d_in[3];
    const float* bhh0 = (const float*)d_in[4];
    const float* Wih1 = (const float*)d_in[5];
    const float* Whh1 = (const float*)d_in[6];
    const float* bih1 = (const float*)d_in[7];
    const float* bhh1 = (const float*)d_in[8];
    const float* Wfc2 = (const float*)d_in[9];
    const float* bfc2 = (const float*)d_in[10];
    const float* Wfc  = (const float*)d_in[11];
    const float* bfc  = (const float*)d_in[12];

    float* hlast = (float*)d_ws;   // 4096 * 32 floats = 512 KB

    gru_seq_kernel<<<BATCH / 16, 64, 0, stream>>>(
        x, Wih0, Whh0, bih0, bhh0, Wih1, Whh1, bih1, bhh1, hlast);

    fc_head_kernel<<<BATCH / 256, 256, 0, stream>>>(
        hlast, Wfc2, bfc2, Wfc, bfc, (float*)d_out);
}
